// NEST2_57501022159147
// MI455X (gfx1250) — matrix operand order repaired and checked by reading, not compiled, not run on hardware
//
#include <hip/hip_runtime.h>
#include <hip/hip_bf16.h>

typedef __attribute__((ext_vector_type(16))) _Float16 v16h;
typedef __attribute__((ext_vector_type(8)))  _Float16 v8h;
typedef __attribute__((ext_vector_type(8)))  float    v8f;
typedef __attribute__((ext_vector_type(4)))  unsigned int v4u;
typedef __attribute__((ext_vector_type(4)))  int      v4i;
typedef __attribute__((ext_vector_type(8)))  int      v8i_t;

#define NMOD_C  4
#define R_C     64
#define K_C     256   // NMOD*R
#define M_C     512
#define B_C     16384
#define LDS_PITCH 264 // halves per A-tile row: 256 + 8 pad (528B = 33*16B, 4-bank skew)

#ifdef __has_builtin
#if __has_builtin(__builtin_amdgcn_tensor_load_to_lds)
#define HAVE_TDM 1
#endif
#endif

#ifndef HAVE_TDM
#warning "CDNA5 TDM builtin __builtin_amdgcn_tensor_load_to_lds NOT available: using cooperative-copy fallback"
#endif

// ---------------------------------------------------------------------------
// Kernel 0: convert S [512,256] f32 -> f16 (rows are K-contiguous = the
// WMMA B-matrix (S^T) per-lane layout).
// ---------------------------------------------------------------------------
__global__ void cvt_s_kernel(const float* __restrict__ S,
                             _Float16* __restrict__ S16, int n) {
    int i = blockIdx.x * blockDim.x + threadIdx.x;
    if (i < n) S16[i] = (_Float16)S[i];
}

// ---------------------------------------------------------------------------
// Kernel 1: stick-breaking embedding. One wave32 per (row, modality).
// Lane l holds elements k=l and k=l+32; ordered exclusive cumsum via wave scan.
// ---------------------------------------------------------------------------
__global__ void __launch_bounds__(256) emb_kernel(
    const int*   __restrict__ idx,
    const float* __restrict__ v0, const float* __restrict__ v1,
    const float* __restrict__ v2, const float* __restrict__ v3,
    _Float16*    __restrict__ emb) {
    const int lane = threadIdx.x & 31;
    const int wid  = threadIdx.x >> 5;
    const int gw   = blockIdx.x * 8 + wid;      // 0 .. B*NMOD-1
    const int row  = gw >> 2;
    const int mod  = gw & 3;

    const float* vt = (mod == 0) ? v0 : (mod == 1) ? v1 : (mod == 2) ? v2 : v3;
    const int id = idx[row * NMOD_C + mod];
    const float* vr = vt + (size_t)id * R_C;

    const float x0 = vr[lane];
    const float x1 = vr[lane + 32];

    const float t0  = log1pf(__expf(-fabsf(x0)));
    const float t1  = log1pf(__expf(-fabsf(x1)));
    const float lv0 = fminf(x0, 0.f) - t0;          // log sigmoid(x0)
    const float lm0 = -fmaxf(x0, 0.f) - t0;         // log sigmoid(-x0)
    const float lv1 = fminf(x1, 0.f) - t1;
    const float lm1 = -fmaxf(x1, 0.f) - t1;

    float s0 = lm0;
    #pragma unroll
    for (int d = 1; d < 32; d <<= 1) {
        float y = __shfl_up(s0, d, 32);
        if (lane >= d) s0 += y;
    }
    const float tot0 = __shfl(s0, 31, 32);

    float s1 = lm1;
    #pragma unroll
    for (int d = 1; d < 32; d <<= 1) {
        float y = __shfl_up(s1, d, 32);
        if (lane >= d) s1 += y;
    }

    const float e0 = __expf(lv0 + (s0 - lm0));
    const float e1 = __expf(lv1 + tot0 + (s1 - lm1));

    _Float16* er = emb + (size_t)row * K_C + mod * R_C;
    er[lane]      = (_Float16)e0;
    er[lane + 32] = (_Float16)e1;
}

// ---------------------------------------------------------------------------
// Kernel 2: TDM-staged A tile in LDS + WMMA GEMM + cos/sin features + matvec
// + sigmoid. One workgroup = 16 batch rows x all 512 columns; 8 waves, each
// wave covers 4 N-tiles of 16 over K=256.
// ---------------------------------------------------------------------------
__device__ __forceinline__ v16h combine16(v8h lo, v8h hi) {
    v16h r;
    #pragma unroll
    for (int i = 0; i < 8; ++i) { r[i] = lo[i]; r[i + 8] = hi[i]; }
    return r;
}

__global__ void __launch_bounds__(256, 2) gemm_fused_kernel(
    const _Float16* __restrict__ emb,   // [B, 256] f16 row-major
    const _Float16* __restrict__ S16,   // [512, 256] f16 row-major (= S^T as B)
    const float*    __restrict__ bias,  // [512]
    const float*    __restrict__ w,     // [1024]
    float*          __restrict__ out) { // [B]
    __shared__ _Float16 atile[16 * LDS_PITCH];  // 16x256 f16 tile, padded rows
    __shared__ float    smem[16];

    const int lane = threadIdx.x & 31;
    const int wid  = threadIdx.x >> 5;
    const int m    = lane & 15;   // N within tile (C/D), row index for A
    const int h    = lane >> 4;   // K-half select for A/B; M+8 select for C/D
    const int row0 = blockIdx.x * 16;

    if (threadIdx.x < 16) smem[threadIdx.x] = 0.f;

#if defined(HAVE_TDM)
    // --- Tensor Data Mover: DMA the 16x256 f16 tile into LDS with row pad ---
    if (wid == 0) {
        const unsigned long long gbase =
            (unsigned long long)(uintptr_t)(emb + (size_t)row0 * K_C);
        const unsigned lds_base =
            (unsigned)(uintptr_t)(void*)atile;   // flat LDS addr low 32 = offset

        v4u g0;
        g0[0] = 1u;                               // count=1, user descriptor
        g0[1] = lds_base;                         // lds_addr
        g0[2] = (unsigned)(gbase & 0xFFFFFFFFu);  // global_addr[31:0]
        g0[3] = (unsigned)((gbase >> 32) & 0x01FFFFFFu) | (2u << 30); // [56:32], type=2

        v8i_t g1;
        g1[0] = (1 << 16)       // data_size = 2 bytes
              | (1 << 20)       // pad_enable
              | (6 << 22)       // pad_interval: 2^(6+1)=128 dwords = one 256-half row
              | (3 << 25);      // pad_amount: 4 dwords = 8 halves
        g1[1] = (K_C & 0xFFFF) << 16;   // tensor_dim0 = 256 (bits 63:48)
        g1[2] = (16 << 16);             // tensor_dim0 hi = 0; tensor_dim1 = 16
        g1[3] = (K_C << 16);            // tensor_dim1 hi = 0; tile_dim0 = 256
        g1[4] = 16;                     // tile_dim1 = 16; tile_dim2 = 0
        g1[5] = K_C;                    // tensor_dim0_stride = 256 (low 32)
        g1[6] = 0;                      // stride0 hi16 = 0; dim1_stride low16 = 0
        g1[7] = 0;

        v4i z4 = {0, 0, 0, 0};
#if __clang_major__ >= 23
        v8i_t z8 = {};
        __builtin_amdgcn_tensor_load_to_lds(g0, g1, z4, z4, z8, 0);
#else
        __builtin_amdgcn_tensor_load_to_lds(g0, g1, z4, z4, 0);
#endif
        __builtin_amdgcn_s_wait_tensorcnt((unsigned short)0);
    }
#else
    // Fallback: cooperative global->LDS copy, 512 chunks of 8 halves (16B)
    for (int c = threadIdx.x; c < 512; c += 256) {
        const int r = c >> 5, q = c & 31;
        *(v8h*)&atile[r * LDS_PITCH + q * 8] =
            *(const v8h*)(emb + (size_t)(row0 + r) * K_C + q * 8);
    }
#endif
    __syncthreads();

    const _Float16* arow = atile + m * LDS_PITCH;   // LDS row for this lane's M

    v8f acc[4];
    #pragma unroll
    for (int t = 0; t < 4; ++t) acc[t] = (v8f){};

    #pragma unroll
    for (int kb = 0; kb < K_C; kb += 32) {
        // A fragment (16x32 f16) from LDS: lane (m,h) takes K=kb+8h+e (e<8)
        // and K=kb+16+8h+(e-8) -> two 16B ds_load_b128 chunks.
        const _Float16* ap = arow + kb + 8 * h;
        const v16h a = combine16(*(const v8h*)ap, *(const v8h*)(ap + 16));

        #pragma unroll
        for (int t = 0; t < 4; ++t) {
            const int n = (wid * 4 + t) * 16 + m;   // column of S^T = row of S
            const _Float16* bp = S16 + (size_t)n * K_C + kb + 8 * h;
            if (kb + 32 < K_C) __builtin_prefetch(bp + 32, 0, 3);
            const v16h bm = combine16(*(const v8h*)bp, *(const v8h*)(bp + 16));
            acc[t] = __builtin_amdgcn_wmma_f32_16x16x32_f16(
                /*neg_a=*/false, a, /*neg_b=*/false, bm,
                /*c_mod=*/(short)0, acc[t],
                /*reuse_a=*/false, /*reuse_b=*/false);
        }
    }

    // Epilogue: acc element r in lane (m,h): M = r + 8h, N = ncol + m.
    float partial[8];
    #pragma unroll
    for (int r = 0; r < 8; ++r) partial[r] = 0.f;

    #pragma unroll
    for (int t = 0; t < 4; ++t) {
        const int n  = (wid * 4 + t) * 16 + m;
        const float bb = bias[n];
        const float wc = w[n];
        const float ws = w[M_C + n];
        #pragma unroll
        for (int r = 0; r < 8; ++r) {
            const float v = acc[t][r] + bb;
            partial[r] += __cosf(v) * wc + __sinf(v) * ws;
        }
    }

    #pragma unroll
    for (int r = 0; r < 8; ++r) {
        float p = partial[r];
        #pragma unroll
        for (int msk = 1; msk < 16; msk <<= 1) p += __shfl_xor(p, msk, 32);
        if (m == 0) atomicAdd(&smem[r + 8 * h], p);   // ds_add_f32 across waves
    }
    __syncthreads();

    if (threadIdx.x < 16) {
        const float s = smem[threadIdx.x];
        out[row0 + threadIdx.x] = 1.f / (1.f + __expf(-s));
    }
}

// ---------------------------------------------------------------------------
// Launch. Inputs: indices, v0, v1, v2, v3, S, b, w_mu. Output: [B] f32.
// Workspace: S16 (256KB) then emb16 (8MB).
// ---------------------------------------------------------------------------
extern "C" void kernel_launch(void* const* d_in, const int* in_sizes, int n_in,
                              void* d_out, int out_size, void* d_ws, size_t ws_size,
                              hipStream_t stream) {
    const int*   indices = (const int*)d_in[0];
    const float* v0 = (const float*)d_in[1];
    const float* v1 = (const float*)d_in[2];
    const float* v2 = (const float*)d_in[3];
    const float* v3 = (const float*)d_in[4];
    const float* S  = (const float*)d_in[5];
    const float* b  = (const float*)d_in[6];
    const float* w  = (const float*)d_in[7];
    float* out = (float*)d_out;

    _Float16* S16 = (_Float16*)d_ws;
    _Float16* emb = (_Float16*)((char*)d_ws + (size_t)M_C * K_C * sizeof(_Float16));

    const int sN = M_C * K_C;  // 131072
    cvt_s_kernel<<<(sN + 255) / 256, 256, 0, stream>>>(S, S16, sN);

    emb_kernel<<<(B_C * NMOD_C) / 8, 256, 0, stream>>>(indices, v0, v1, v2, v3, emb);

    gemm_fused_kernel<<<B_C / 16, 256, 0, stream>>>(emb, S16, b, w, out);
}
